// Main_Net_79852031967513
// MI455X (gfx1250) — matrix operand order repaired and checked by reading, not compiled, not run on hardware
//
#include <hip/hip_runtime.h>

// ---------------------------------------------------------------------------
// MI455X (gfx1250) implementation.
//
// out[b,c] = sum_{j,i} f1[b,j] * f0[b,i] * W_out[j*256+i, c] + b_out[c]
//          = < F1[b,:], (F0 @ M_c^T)[b,:] > + b_out[c],  M_c = W_out[:,c].reshape(256,256)
//
// Pipeline (all bf16x3 split-precision WMMA, f32 accumulate):
//   GEMM1: H_z  = relu(x_z @ W_feat + b_feat)        [2048,512]   z = 0,1
//   GEMM2: F_z  = H_z @ W_br_z + b_br_z              [2048,256]
//   GEMM3: T_c  = F0 @ M_c^T                         [2048,256]   c = 0,1
//   REDUCE: out[b,c] = dot(F1[b,:], T_c[b,:]) + b_out[c]
// ---------------------------------------------------------------------------

typedef __attribute__((ext_vector_type(16))) __bf16        v16bf;
typedef __attribute__((ext_vector_type(8)))  float         v8f;
typedef __attribute__((ext_vector_type(4)))  unsigned int  u32x4;
typedef __attribute__((ext_vector_type(4)))  float         f32x4;

union Pack8 { unsigned short s[8]; u32x4 u; };
union Frag  { v16bf bf; u32x4 u[2]; };

__device__ __forceinline__ unsigned short f32_to_bf16_rne(float f) {
    unsigned u = __float_as_uint(f);
    unsigned r = u + 0x7FFFu + ((u >> 16) & 1u);
    return (unsigned short)(r >> 16);
}
__device__ __forceinline__ float bf16_bits_to_f32(unsigned short h) {
    return __uint_as_float(((unsigned)h) << 16);
}

#define LDSTRIDE 40  // shorts per LDS row (80B, padded: 20 banks/row -> conflict-free)

__device__ __forceinline__ void store_ctile(float* __restrict__ C, v8f acc, int N,
                                            int m_base, int n, const float* __restrict__ bias,
                                            int epilogue) {
    float bv = (epilogue > 0) ? bias[n] : 0.0f;
#pragma unroll
    for (int v = 0; v < 8; ++v) {
        float val = acc[v] + bv;
        if (epilogue == 2) val = fmaxf(val, 0.0f);
        C[(size_t)(m_base + v) * N + n] = val;
    }
}

// Generic tiled GEMM: C = epilogue(A[M,K] @ B[K,N]); B element (k,n) at
// Bsrc[k*bsk + n*bsn]. blockIdx.z selects the {A,B,bias,C} pair.
// epilogue: 0 = none, 1 = +bias, 2 = relu(+bias)
__global__ __launch_bounds__(256) void gemm_bf16x3_kernel(
    const float* __restrict__ A0, const float* __restrict__ A1,
    const float* __restrict__ B0, const float* __restrict__ B1,
    const float* __restrict__ bias0, const float* __restrict__ bias1,
    float* __restrict__ C0, float* __restrict__ C1,
    int M, int N, int K, int bsk, int bsn, int epilogue)
{
    const int z = blockIdx.z;
    const float* __restrict__ A    = z ? A1 : A0;
    const float* __restrict__ B    = z ? B1 : B0;
    const float* __restrict__ bias = z ? bias1 : bias0;
    float* __restrict__ C          = z ? C1 : C0;

    const int m0 = blockIdx.x * 64;
    const int n0 = blockIdx.y * 64;

    __shared__ unsigned short Ah[64 * LDSTRIDE];
    __shared__ unsigned short Al[64 * LDSTRIDE];
    __shared__ unsigned short Bh[64 * LDSTRIDE];   // staged col-major: [n][k]
    __shared__ unsigned short Bl[64 * LDSTRIDE];

    const int tid  = threadIdx.x;
    const int wid  = tid >> 5;       // 8 waves
    const int lane = tid & 31;
    const int wm   = wid & 3;        // 4 wave-rows of 16 M
    const int wn   = wid >> 2;       // 2 wave-cols of 32 N
    const int lm   = lane & 15;
    const int lh   = lane >> 4;      // lane half selects K sub-range per ISA layout

    // cooperative staging assignments (256 threads, 2048 elems per tile)
    const int arow = tid >> 2;        // 0..63
    const int akq  = tid & 3;         // 8-float quarter of the K=32 row
    const int bn   = tid & 63;        // 0..63
    const int bk0  = (tid >> 6) * 8;  // 0,8,16,24

    v8f acc0 = {0.f,0.f,0.f,0.f,0.f,0.f,0.f,0.f};
    v8f acc1 = {0.f,0.f,0.f,0.f,0.f,0.f,0.f,0.f};

    const int ksteps = K >> 5;
    for (int kt = 0; kt < ksteps; ++kt) {
        const int k0 = kt << 5;

        // ---- stage A tile: 64x32 f32 -> bf16 hi/lo planes ----
        {
            const float* ap = A + (size_t)(m0 + arow) * K + (k0 + akq * 8);
            f32x4 a0v = *(const f32x4*)ap;
            f32x4 a1v = *(const f32x4*)(ap + 4);
            if (kt + 1 < ksteps)
                __builtin_prefetch((const void*)(ap + 32), 0, 0);  // next K-step tile
            float v[8] = {a0v.x, a0v.y, a0v.z, a0v.w, a1v.x, a1v.y, a1v.z, a1v.w};
            Pack8 hi, lo;
#pragma unroll
            for (int j = 0; j < 8; ++j) {
                unsigned short h = f32_to_bf16_rne(v[j]);
                hi.s[j] = h;
                lo.s[j] = f32_to_bf16_rne(v[j] - bf16_bits_to_f32(h));
            }
            const int off = arow * LDSTRIDE + akq * 8;
            *(u32x4*)&Ah[off] = hi.u;
            *(u32x4*)&Al[off] = lo.u;
        }
        // ---- stage B tile: 32x64 (generic strides) -> col-major bf16 hi/lo ----
        {
            Pack8 hi, lo;
#pragma unroll
            for (int j = 0; j < 8; ++j) {
                float v = B[(size_t)(k0 + bk0 + j) * bsk + (size_t)(n0 + bn) * bsn];
                unsigned short h = f32_to_bf16_rne(v);
                hi.s[j] = h;
                lo.s[j] = f32_to_bf16_rne(v - bf16_bits_to_f32(h));
            }
            const int off = bn * LDSTRIDE + bk0;
            *(u32x4*)&Bh[off] = hi.u;
            *(u32x4*)&Bl[off] = lo.u;
        }
        __syncthreads();

        // ---- fragments (b128 LDS loads matching ISA WMMA VGPR layouts) ----
        // A 16x32: lanes<16 K{0..7,16..23}, lanes>=16 K{8..15,24..31}, M=lane%16
        Frag fah, fal;
        {
            const unsigned short* ah = &Ah[(wm * 16 + lm) * LDSTRIDE];
            const unsigned short* al = &Al[(wm * 16 + lm) * LDSTRIDE];
            fah.u[0] = *(const u32x4*)(ah + lh * 8);
            fah.u[1] = *(const u32x4*)(ah + 16 + lh * 8);
            fal.u[0] = *(const u32x4*)(al + lh * 8);
            fal.u[1] = *(const u32x4*)(al + 16 + lh * 8);
        }
        // B 32x16: lanes<16 K{0..7,16..23}, lanes>=16 K{8..15,24..31}, N=lane%16
        Frag fbh0, fbl0, fbh1, fbl1;
        {
            const unsigned short* p;
            p = &Bh[(wn * 32 + lm) * LDSTRIDE + lh * 8];
            fbh0.u[0] = *(const u32x4*)p;  fbh0.u[1] = *(const u32x4*)(p + 16);
            p = &Bl[(wn * 32 + lm) * LDSTRIDE + lh * 8];
            fbl0.u[0] = *(const u32x4*)p;  fbl0.u[1] = *(const u32x4*)(p + 16);
            p = &Bh[(wn * 32 + 16 + lm) * LDSTRIDE + lh * 8];
            fbh1.u[0] = *(const u32x4*)p;  fbh1.u[1] = *(const u32x4*)(p + 16);
            p = &Bl[(wn * 32 + 16 + lm) * LDSTRIDE + lh * 8];
            fbl1.u[0] = *(const u32x4*)p;  fbl1.u[1] = *(const u32x4*)(p + 16);
        }

        // ---- bf16x3: A*B ~= Ah*Bh + Ah*Bl + Al*Bh, f32 accumulate ----
        acc0 = __builtin_amdgcn_wmma_f32_16x16x32_bf16(false, fah.bf, false, fbh0.bf, (short)0, acc0, false, false);
        acc0 = __builtin_amdgcn_wmma_f32_16x16x32_bf16(false, fah.bf, false, fbl0.bf, (short)0, acc0, false, false);
        acc0 = __builtin_amdgcn_wmma_f32_16x16x32_bf16(false, fal.bf, false, fbh0.bf, (short)0, acc0, false, false);
        acc1 = __builtin_amdgcn_wmma_f32_16x16x32_bf16(false, fah.bf, false, fbh1.bf, (short)0, acc1, false, false);
        acc1 = __builtin_amdgcn_wmma_f32_16x16x32_bf16(false, fah.bf, false, fbl1.bf, (short)0, acc1, false, false);
        acc1 = __builtin_amdgcn_wmma_f32_16x16x32_bf16(false, fal.bf, false, fbh1.bf, (short)0, acc1, false, false);

        __syncthreads();
    }

    // ---- epilogue: C 16x16 f32 layout -> lane(N=lm), vgpr v (M = lh*8 + v) ----
    const int m_base = m0 + wm * 16 + lh * 8;
    store_ctile(C, acc0, N, m_base, n0 + wn * 32 + lm,      bias, epilogue);
    store_ctile(C, acc1, N, m_base, n0 + wn * 32 + 16 + lm, bias, epilogue);
}

// out[b,c] = dot(F1[b,:], T_c[b,:]) + b_out[c]   (one block per batch row)
__global__ __launch_bounds__(256) void kron_reduce_kernel(
    const float* __restrict__ F1, const float* __restrict__ T0,
    const float* __restrict__ T1, const float* __restrict__ b_out,
    float* __restrict__ out)
{
    __shared__ float s0[256];
    __shared__ float s1[256];
    const int b = blockIdx.x;
    const int j = threadIdx.x;
    const float p = F1[(size_t)b * 256 + j];
    s0[j] = p * T0[(size_t)b * 256 + j];
    s1[j] = p * T1[(size_t)b * 256 + j];
    __syncthreads();
#pragma unroll
    for (int off = 128; off > 0; off >>= 1) {
        if (j < off) { s0[j] += s0[j + off]; s1[j] += s1[j + off]; }
        __syncthreads();
    }
    if (j == 0) {
        out[(size_t)b * 2 + 0] = s0[0] + b_out[0];
        out[(size_t)b * 2 + 1] = s1[0] + b_out[1];
    }
}

extern "C" void kernel_launch(void* const* d_in, const int* in_sizes, int n_in,
                              void* d_out, int out_size, void* d_ws, size_t ws_size,
                              hipStream_t stream) {
    (void)in_sizes; (void)n_in; (void)out_size; (void)ws_size;
    constexpr int Bsz = 2048, IN_DIM = 1024, HID = 512, FEAT = 256;

    const float* x0     = (const float*)d_in[0];
    const float* x1     = (const float*)d_in[1];
    const float* W_feat = (const float*)d_in[2];
    const float* b_feat = (const float*)d_in[3];
    const float* W_br0  = (const float*)d_in[4];
    const float* b_br0  = (const float*)d_in[5];
    const float* W_br1  = (const float*)d_in[6];
    const float* b_br1  = (const float*)d_in[7];
    const float* W_out  = (const float*)d_in[8];
    const float* b_out  = (const float*)d_in[9];
    float* out = (float*)d_out;

    // workspace: H(2*2048*512) | F0(2048*256) | F1(2048*256) | T(2*2048*256) = 16 MB
    float* ws = (float*)d_ws;
    float* H  = ws;
    float* H0 = H;
    float* H1 = H + (size_t)Bsz * HID;
    float* F0 = H + (size_t)2 * Bsz * HID;
    float* F1 = F0 + (size_t)Bsz * FEAT;
    float* T0 = F1 + (size_t)Bsz * FEAT;
    float* T1 = T0 + (size_t)Bsz * FEAT;

    const dim3 blk(256);

    // GEMM1: H_z = relu(x_z @ W_feat + b_feat)  [2048,1024]x[1024,512]
    gemm_bf16x3_kernel<<<dim3(Bsz / 64, HID / 64, 2), blk, 0, stream>>>(
        x0, x1, W_feat, W_feat, b_feat, b_feat, H0, H1,
        Bsz, HID, IN_DIM, /*bsk=*/HID, /*bsn=*/1, /*epilogue=*/2);

    // GEMM2: F_z = H_z @ W_br_z + b_br_z        [2048,512]x[512,256]
    gemm_bf16x3_kernel<<<dim3(Bsz / 64, FEAT / 64, 2), blk, 0, stream>>>(
        H0, H1, W_br0, W_br1, b_br0, b_br1, F0, F1,
        Bsz, FEAT, HID, /*bsk=*/FEAT, /*bsn=*/1, /*epilogue=*/1);

    // GEMM3: T_c = F0 @ M_c^T,  M_c(j,i) = W_out[(j*256+i)*2 + c]
    //   => B element (k=i, n=j) at (W_out + c)[i*2 + j*512]
    gemm_bf16x3_kernel<<<dim3(Bsz / 64, FEAT / 64, 2), blk, 0, stream>>>(
        F0, F0, W_out + 0, W_out + 1, nullptr, nullptr, T0, T1,
        Bsz, FEAT, FEAT, /*bsk=*/2, /*bsn=*/FEAT * 2, /*epilogue=*/0);

    // out[b,c] = dot(F1[b,:], T_c[b,:]) + b_out[c]
    kron_reduce_kernel<<<dim3(Bsz), blk, 0, stream>>>(F1, T0, T1, b_out, out);
}
